// AttentionBlock_70712341561783
// MI455X (gfx1250) — matrix operand order repaired and checked by reading, not compiled, not run on hardware
//
#include <hip/hip_runtime.h>

// ---------------------------------------------------------------------------
// CDNA5 (gfx1250, wave32) causal attention block:
//   K = X Wk + bk ; Q = X Wq + bq ; V = X Wv + bv
//   S = Q K^T / sqrt(1024) ; causal mask ; P = softmax(S) ; read = P V
//   out = concat(X, read)
// GEMMs: async global->LDS staging (ASYNCcnt, double-buffered) feeding
//        v_wmma_f32_16x16x32_bf16 from ds_load_b128 fragments.
// ---------------------------------------------------------------------------

typedef __attribute__((ext_vector_type(16))) __bf16 v16bf;
typedef __attribute__((ext_vector_type(8)))  __bf16 v8bf;
typedef __attribute__((ext_vector_type(8)))  float  v8f;

typedef __attribute__((address_space(3))) __bf16 lds_bf16;

// Exact pointee type expected by __builtin_amdgcn_global_load_async_to_lds_b128
// (from clang diagnostic: "int __vector(4) __device__ *").
typedef int v4i __attribute__((vector_size(16)));
typedef __attribute__((address_space(1))) v4i as1_v4i;
typedef __attribute__((address_space(3))) v4i as3_v4i;

#define BATCH 4
#define SEQ   2048
#define DIM   1024
#define ROWS  (BATCH * SEQ)          // 8192
#define NEG_FILL (-1.0e9f)
#define SM_SCALE (0.03125f)          // 1/sqrt(1024)

#define KTILE 32
#define TILE_ELEMS (128 * KTILE)     // one 128x32 bf16 tile = 8KB

// ---------------------------------------------------------------------------
// Async global->LDS copy (16B) + ASYNCcnt wait
// ---------------------------------------------------------------------------
__device__ __forceinline__ void async_copy_b128(lds_bf16* l, const __bf16* g) {
#if __has_builtin(__builtin_amdgcn_global_load_async_to_lds_b128)
  void* gg = const_cast<__bf16*>(g);
  __attribute__((address_space(1))) void* ga =
      (__attribute__((address_space(1))) void*)gg;
  __attribute__((address_space(3))) void* la =
      (__attribute__((address_space(3))) void*)l;
  __builtin_amdgcn_global_load_async_to_lds_b128(
      (as1_v4i*)ga, (as3_v4i*)la, 0, 0);
#else
  asm volatile("global_load_async_to_lds_b128 %0, %1, off"
               :: "v"(l), "v"(g)
               : "memory");
#endif
}

template <int N>
__device__ __forceinline__ void wait_asynccnt() {
#if __has_builtin(__builtin_amdgcn_s_wait_asynccnt)
  __builtin_amdgcn_s_wait_asynccnt(N);
#else
  asm volatile("s_wait_asynccnt %0" :: "n"(N) : "memory");
#endif
}

// Stage one 128x32 bf16 tile (rows r of base, cols k0..k0+31) into LDS.
// 256 threads x 2 async b128 ops (16 bf16 per thread).
__device__ __forceinline__ void stage_tile(lds_bf16* dst, const __bf16* __restrict__ base,
                                           int ld, int k0) {
  const int tid = threadIdx.x;
  const int c  = (tid & 3) * 8;      // 0,8,16,24
  const int r0 = tid >> 2;           // 0..63
#pragma unroll
  for (int chunk = 0; chunk < 2; ++chunk) {
    const int r = chunk * 64 + r0;   // 0..127
    async_copy_b128(dst + (size_t)r * KTILE + c,
                    base + (size_t)r * ld + k0 + c);
  }
}

// ---------------------------------------------------------------------------
// Fragment helpers (ISA 7.12.2 layouts, wave32). Sources live in LDS (ld=32)
// so these lower to ds_load_b128.
// ---------------------------------------------------------------------------
__device__ __forceinline__ v8f zero8() {
  v8f v = {0.f, 0.f, 0.f, 0.f, 0.f, 0.f, 0.f, 0.f};
  return v;
}

// A 16x32 fragment: lane L: row m=L%16, half h=L/16; elems 0..7 <- K=h*8..h*8+7,
// elems 8..15 <- K=16+h*8..16+h*8+7.
__device__ __forceinline__ v16bf load_frag_a(const __bf16* tile, int lda) {
  const int lane = threadIdx.x & 31;
  const int m = lane & 15, h = lane >> 4;
  const __bf16* p = tile + (size_t)m * lda + h * 8;
  union { v16bf w; v8bf hv[2]; } u;
  u.hv[0] = *(const v8bf*)(p);
  u.hv[1] = *(const v8bf*)(p + 16);
  return u.w;
}

// B 32x16 fragment from [n][k] storage: lane L: n=L%16, h=L/16; elems 0..7 <- K=h*8..h*8+7,
// elems 8..15 <- K=16+h*8..16+h*8+7.
__device__ __forceinline__ v16bf load_frag_b(const __bf16* tile, int ldb) {
  const int lane = threadIdx.x & 31;
  const int n = lane & 15, h = lane >> 4;
  const __bf16* p = tile + (size_t)n * ldb + h * 8;
  union { v16bf w; v8bf hv[2]; } u;
  u.hv[0] = *(const v8bf*)(p);
  u.hv[1] = *(const v8bf*)(p + 16);
  return u.w;
}

__device__ __forceinline__ void store_tile_f32(float* __restrict__ out, int ld, v8f c) {
  const int lane = threadIdx.x & 31;
  const int n = lane & 15, h = lane >> 4;
#pragma unroll
  for (int r = 0; r < 8; ++r) out[(size_t)(h * 8 + r) * ld + n] = c[r];
}

// One K-step of the 64x32 wave tile: 4x2 accumulators, 8 wmma.
__device__ __forceinline__ void wave_mma(v8f acc[4][2],
                                         const __bf16* A, const __bf16* Bm) {
  v16bf a[4], b[2];
#pragma unroll
  for (int i = 0; i < 4; ++i) a[i] = load_frag_a(A + (size_t)i * 16 * KTILE, KTILE);
#pragma unroll
  for (int j = 0; j < 2; ++j) b[j] = load_frag_b(Bm + (size_t)j * 16 * KTILE, KTILE);
#pragma unroll
  for (int i = 0; i < 4; ++i)
#pragma unroll
    for (int j = 0; j < 2; ++j)
      acc[i][j] = __builtin_amdgcn_wmma_f32_16x16x32_bf16(
          false, a[i], false, b[j], (short)0, acc[i][j], false, false);
}

// Double-buffered LDS GEMM body: block tile 128x128, wave tile 64x32.
// Abase points at (block row 0, k=0); Bbase at (block col-row 0, k=0).
__device__ __forceinline__ void gemm_lds(v8f acc[4][2],
                                         const __bf16* __restrict__ Abase, int lda,
                                         const __bf16* __restrict__ Bbase, int ldb,
                                         int kmax,
                                         __bf16 (*As)[TILE_ELEMS],
                                         __bf16 (*Bs)[TILE_ELEMS],
                                         int wm, int wn) {
  lds_bf16* Asl[2] = {(lds_bf16*)&As[0][0], (lds_bf16*)&As[1][0]};
  lds_bf16* Bsl[2] = {(lds_bf16*)&Bs[0][0], (lds_bf16*)&Bs[1][0]};

  const int nk = kmax >> 5;          // K-steps of 32
  stage_tile(Asl[0], Abase, lda, 0);
  stage_tile(Bsl[0], Bbase, ldb, 0);

  for (int kk = 0; kk < nk; ++kk) {
    const int cur = kk & 1;
    if (kk + 1 < nk) {
      stage_tile(Asl[cur ^ 1], Abase, lda, (kk + 1) * KTILE);
      stage_tile(Bsl[cur ^ 1], Bbase, ldb, (kk + 1) * KTILE);
      wait_asynccnt<4>();            // 4 ops of the next stage may stay in flight
    } else {
      wait_asynccnt<0>();
    }
    __syncthreads();                 // stage kk visible to whole workgroup
    wave_mma(acc, &As[cur][(size_t)wm * KTILE], &Bs[cur][(size_t)wn * KTILE]);
    __syncthreads();                 // done reading before buffer is re-staged
  }
}

// ---------------------------------------------------------------------------
// Kernel 1: X fp32 -> bf16, and copy X into out[..., 0:1024]
// ---------------------------------------------------------------------------
__global__ void __launch_bounds__(256) convert_x_kernel(const float* __restrict__ X,
                                                        __bf16* __restrict__ Xb,
                                                        float* __restrict__ out) {
  const int i = blockIdx.x * 256 + threadIdx.x;     // ROWS*DIM threads
  const float v = X[i];
  Xb[i] = (__bf16)v;
  const int row = i >> 10, col = i & 1023;
  out[(size_t)row * (2 * DIM) + col] = v;
}

// ---------------------------------------------------------------------------
// Kernel 2: transpose Wq/Wk/Wv (fp32 [d][n]) -> bf16 Wt[z][n][d]
// ---------------------------------------------------------------------------
__global__ void __launch_bounds__(256) convert_w_kernel(const float* __restrict__ Wq,
                                                        const float* __restrict__ Wk,
                                                        const float* __restrict__ Wv,
                                                        __bf16* __restrict__ Wt) {
  const int i = blockIdx.x * 256 + threadIdx.x;     // 3*1024*1024 threads
  const int z = i >> 20;
  const int r = i & ((1 << 20) - 1);
  const int n = r >> 10, d = r & 1023;
  const float* __restrict__ W = (z == 0) ? Wq : (z == 1) ? Wk : Wv;
  Wt[i] = (__bf16)W[(size_t)d * DIM + n];
}

// ---------------------------------------------------------------------------
// Kernel 3: QKV projection GEMM. grid = (ROWS/128, DIM/128, 3)
//   z=0 -> Q (row-major bf16), z=1 -> K (row-major bf16), z=2 -> V transposed
// ---------------------------------------------------------------------------
__global__ void __launch_bounds__(256) qkv_gemm_kernel(const __bf16* __restrict__ Xb,
                                                       const __bf16* __restrict__ Wt,
                                                       const float* __restrict__ bq,
                                                       const float* __restrict__ bk,
                                                       const float* __restrict__ bv,
                                                       __bf16* __restrict__ Q,
                                                       __bf16* __restrict__ K,
                                                       __bf16* __restrict__ Vt) {
  __shared__ __bf16 As[2][TILE_ELEMS];
  __shared__ __bf16 Bs[2][TILE_ELEMS];

  const int z  = blockIdx.z;
  const int m0 = blockIdx.x * 128;
  const int n0 = blockIdx.y * 128;
  const int wid = threadIdx.x >> 5;
  const int wm = (wid >> 2) * 64;
  const int wn = (wid & 3) * 32;

  const __bf16* __restrict__ W    = Wt + (size_t)z * DIM * DIM;
  const float*  __restrict__ bias = (z == 0) ? bq : (z == 1) ? bk : bv;

  v8f acc[4][2];
#pragma unroll
  for (int i = 0; i < 4; ++i)
#pragma unroll
    for (int j = 0; j < 2; ++j) acc[i][j] = zero8();

  gemm_lds(acc, Xb + (size_t)m0 * DIM, DIM, W + (size_t)n0 * DIM, DIM,
           DIM, As, Bs, wm, wn);

  const int lane = threadIdx.x & 31;
  const int cn = lane & 15, h = lane >> 4;
#pragma unroll
  for (int i = 0; i < 4; ++i) {
#pragma unroll
    for (int j = 0; j < 2; ++j) {
      const int gn = n0 + wn + j * 16 + cn;                  // output column
      const float badd = bias[gn];
      const int gm = m0 + wm + i * 16;                       // output row base
      if (z == 2) {
        // V transposed: Vt[b][v][t], t contiguous (B-operand of P*V)
        const int b = gm / SEQ;
        const int t = gm & (SEQ - 1);
        __bf16* dst = Vt + ((size_t)b * DIM + gn) * SEQ + t;
#pragma unroll
        for (int r = 0; r < 8; ++r) dst[h * 8 + r] = (__bf16)(acc[i][j][r] + badd);
      } else {
        __bf16* dst = ((z == 0) ? Q : K) + (size_t)gm * DIM + n0 + wn + j * 16;
#pragma unroll
        for (int r = 0; r < 8; ++r)
          dst[(size_t)(h * 8 + r) * DIM + cn] = (__bf16)(acc[i][j][r] + badd);
      }
    }
  }
}

// ---------------------------------------------------------------------------
// Kernel 4: S = Q K^T (raw, unscaled). grid = (SEQ/128, SEQ/128, BATCH)
// Tiles fully above the causal diagonal are skipped (softmax masks them).
// ---------------------------------------------------------------------------
__global__ void __launch_bounds__(256) scores_kernel(const __bf16* __restrict__ Q,
                                                     const __bf16* __restrict__ K,
                                                     float* __restrict__ S) {
  __shared__ __bf16 As[2][TILE_ELEMS];
  __shared__ __bf16 Bs[2][TILE_ELEMS];

  const int b  = blockIdx.z;
  const int q0 = blockIdx.x * 128;
  const int s0 = blockIdx.y * 128;
  if (s0 > q0 + 127) return;                 // entirely masked tile

  const int wid = threadIdx.x >> 5;
  const int wm = (wid >> 2) * 64;
  const int wn = (wid & 3) * 32;

  v8f acc[4][2];
#pragma unroll
  for (int i = 0; i < 4; ++i)
#pragma unroll
    for (int j = 0; j < 2; ++j) acc[i][j] = zero8();

  gemm_lds(acc,
           Q + ((size_t)b * SEQ + q0) * DIM, DIM,
           K + ((size_t)b * SEQ + s0) * DIM, DIM,
           DIM, As, Bs, wm, wn);

  float* Sb = S + ((size_t)b * SEQ + q0 + wm) * SEQ + s0 + wn;
#pragma unroll
  for (int i = 0; i < 4; ++i)
#pragma unroll
    for (int j = 0; j < 2; ++j)
      store_tile_f32(Sb + (size_t)i * 16 * SEQ + j * 16, SEQ, acc[i][j]);
}

// ---------------------------------------------------------------------------
// Kernel 5: causal softmax over each row; scale + (-1e9) mask applied here.
// One 256-thread block per row (8 elements/thread); wave32 shuffle + LDS reduce.
// ---------------------------------------------------------------------------
__global__ void __launch_bounds__(256) softmax_kernel(const float* __restrict__ S,
                                                      __bf16* __restrict__ P) {
  const int row = blockIdx.x;                // 0 .. ROWS-1
  const int q   = row & (SEQ - 1);
  const float* __restrict__ src = S + (size_t)row * SEQ;
  __bf16* __restrict__ dst      = P + (size_t)row * SEQ;

  __shared__ float red[8];
  const int tid = threadIdx.x;
  const int wid = tid >> 5;

  float x[8];
  float m = -3.4e38f;
#pragma unroll
  for (int i = 0; i < 8; ++i) {
    const int s = tid + i * 256;
    float v = src[s] * SM_SCALE;
    v = (s <= q) ? v : NEG_FILL;
    x[i] = v;
    m = fmaxf(m, v);
  }
#pragma unroll
  for (int off = 16; off > 0; off >>= 1) m = fmaxf(m, __shfl_xor(m, off, 32));
  if ((tid & 31) == 0) red[wid] = m;
  __syncthreads();
#pragma unroll
  for (int w = 0; w < 8; ++w) m = fmaxf(m, red[w]);
  __syncthreads();

  float sum = 0.f;
#pragma unroll
  for (int i = 0; i < 8; ++i) {
    x[i] = __expf(x[i] - m);
    sum += x[i];
  }
#pragma unroll
  for (int off = 16; off > 0; off >>= 1) sum += __shfl_xor(sum, off, 32);
  if ((tid & 31) == 0) red[wid] = sum;
  __syncthreads();
  float tot = 0.f;
#pragma unroll
  for (int w = 0; w < 8; ++w) tot += red[w];

  const float inv = 1.0f / tot;
#pragma unroll
  for (int i = 0; i < 8; ++i) dst[tid + i * 256] = (__bf16)(x[i] * inv);
}

// ---------------------------------------------------------------------------
// Kernel 6: read = P * V, written to out[..., 1024:2048].
// grid = (SEQ/128, DIM/128, BATCH); K-loop clipped at q0+128 (P==0 above diag).
// ---------------------------------------------------------------------------
__global__ void __launch_bounds__(256) out_gemm_kernel(const __bf16* __restrict__ P,
                                                       const __bf16* __restrict__ Vt,
                                                       float* __restrict__ out) {
  __shared__ __bf16 As[2][TILE_ELEMS];
  __shared__ __bf16 Bs[2][TILE_ELEMS];

  const int b  = blockIdx.z;
  const int q0 = blockIdx.x * 128;
  const int v0 = blockIdx.y * 128;
  const int wid = threadIdx.x >> 5;
  const int wm = (wid >> 2) * 64;
  const int wn = (wid & 3) * 32;

  v8f acc[4][2];
#pragma unroll
  for (int i = 0; i < 4; ++i)
#pragma unroll
    for (int j = 0; j < 2; ++j) acc[i][j] = zero8();

  const int kmax = (q0 + 128 < SEQ) ? (q0 + 128) : SEQ;
  gemm_lds(acc,
           P  + ((size_t)b * SEQ + q0) * SEQ, SEQ,
           Vt + ((size_t)b * DIM + v0) * SEQ, SEQ,
           kmax, As, Bs, wm, wn);

  float* O = out + ((size_t)b * SEQ + q0 + wm) * (2 * DIM) + DIM + v0 + wn;
#pragma unroll
  for (int i = 0; i < 4; ++i)
#pragma unroll
    for (int j = 0; j < 2; ++j)
      store_tile_f32(O + (size_t)i * 16 * (2 * DIM) + j * 16, 2 * DIM, acc[i][j]);
}

// ---------------------------------------------------------------------------
// Host launch. d_in order: inputs, Wk, bk, Wq, bq, Wv, bv.
// Workspace layout (bytes):
//   Xb 0..16M | Qb 16M | Kb 32M | Vt 48M | Wt 64M (6M) | S 70M (64M) | P 134M (32M)
// ---------------------------------------------------------------------------
extern "C" void kernel_launch(void* const* d_in, const int* in_sizes, int n_in,
                              void* d_out, int out_size, void* d_ws, size_t ws_size,
                              hipStream_t stream) {
  (void)in_sizes; (void)n_in; (void)out_size; (void)ws_size;

  const float* X  = (const float*)d_in[0];
  const float* Wk = (const float*)d_in[1];
  const float* bk = (const float*)d_in[2];
  const float* Wq = (const float*)d_in[3];
  const float* bq = (const float*)d_in[4];
  const float* Wv = (const float*)d_in[5];
  const float* bv = (const float*)d_in[6];
  float* out = (float*)d_out;

  char* ws = (char*)d_ws;
  __bf16* Xb = (__bf16*)(ws + (size_t)0);
  __bf16* Qb = (__bf16*)(ws + ((size_t)16 << 20));
  __bf16* Kb = (__bf16*)(ws + ((size_t)32 << 20));
  __bf16* Vt = (__bf16*)(ws + ((size_t)48 << 20));
  __bf16* Wt = (__bf16*)(ws + ((size_t)64 << 20));   // [Wq, Wk, Wv] transposed
  float*  S  = (float*) (ws + ((size_t)70 << 20));
  __bf16* P  = (__bf16*)(ws + ((size_t)134 << 20));

  // 1) X -> bf16 and copy into out[:, :, 0:1024]
  convert_x_kernel<<<(ROWS * DIM) / 256, 256, 0, stream>>>(X, Xb, out);

  // 2) W -> bf16 transposed (order: Wq, Wk, Wv)
  convert_w_kernel<<<(3 * DIM * DIM) / 256, 256, 0, stream>>>(Wq, Wk, Wv, Wt);

  // 3) Q/K/V projection (z: 0=Q, 1=K, 2=V-transposed)
  qkv_gemm_kernel<<<dim3(ROWS / 128, DIM / 128, 3), 256, 0, stream>>>(
      Xb, Wt, bq, bk, bv, Qb, Kb, Vt);

  // 4) S = Q K^T (causal tiles only)
  scores_kernel<<<dim3(SEQ / 128, SEQ / 128, BATCH), 256, 0, stream>>>(Qb, Kb, S);

  // 5) softmax with scale + causal mask -> P (bf16)
  softmax_kernel<<<ROWS, 256, 0, stream>>>(S, P);

  // 6) read = P V -> out[:, :, 1024:2048]
  out_gemm_kernel<<<dim3(SEQ / 128, DIM / 128, BATCH), 256, 0, stream>>>(P, Vt, out);
}
